// DecorrelatedBatchNorm_32160715112571
// MI455X (gfx1250) — matrix operand order repaired and checked by reading, not compiled, not run on hardware
//
#include <hip/hip_runtime.h>
#include <math.h>

// CDNA5 wave32 WMMA vector types
typedef __attribute__((ext_vector_type(16))) __bf16 v16bf;
typedef __attribute__((ext_vector_type(8)))  __bf16 v8bf;
typedef __attribute__((ext_vector_type(8)))  float  v8f;

#define C_DIM 256
#define EPS_F 0.001f

__device__ __forceinline__ v16bf cat16(v8bf lo, v8bf hi) {
  return __builtin_shufflevector(lo, hi, 0, 1, 2, 3, 4, 5, 6, 7,
                                         8, 9, 10, 11, 12, 13, 14, 15);
}

__device__ __forceinline__ v8f wmma_bf16(v16bf a, v16bf b, v8f c) {
  return __builtin_amdgcn_wmma_f32_16x16x32_bf16(false, a, false, b,
                                                 (short)0, c, false, false);
}

// ---------------------------------------------------------------------------
// Pass 1: per-channel sums (NHWC: channel contiguous -> fully coalesced).
// ---------------------------------------------------------------------------
__global__ __launch_bounds__(256) void channel_sum_kernel(
    const float* __restrict__ x, float* __restrict__ sums) {
  const int t = threadIdx.x;
  const long r0 = (long)blockIdx.x * 256;
  float acc = 0.f;
  for (int r = 0; r < 256; ++r) acc += x[(r0 + r) * C_DIM + t];
  atomicAdd(&sums[t], acc);
}

__global__ __launch_bounds__(256) void finalize_mean_kernel(
    float* __restrict__ mean, float inv_n) {
  mean[threadIdx.x] *= inv_n;
}

// ---------------------------------------------------------------------------
// Pass 2: raw second moment  cov_raw = sum_k x[k] x[k]^T  (mean fixed later).
// Block = 256 threads (8 waves). One channel-major LDS tile Cs[256][k=32]
// serves BOTH WMMA operands: wave w owns ci stripe (blockIdx.x*8+w)*16 and
// sweeps all 16 cj groups -> 128 v_wmma per staged tile. x is read only 2x.
// Operand slices are contiguous 16B v8bf LDS loads (no scalar gathers).
// ---------------------------------------------------------------------------
__global__ __launch_bounds__(256) void cov_kernel(
    const float* __restrict__ x, float* __restrict__ cov, int ksteps) {
  __shared__ __bf16 Cs[C_DIM][40];   // [channel][k] ; 80B row stride (16B mult)

  const int tid = threadIdx.x;
  const int wave = tid >> 5, lane = tid & 31;
  const int half = lane >> 4, m = lane & 15;
  const int ci = ((blockIdx.x << 3) + wave) << 4;       // this wave's ci stripe
  const long rowbase = (long)blockIdx.y * ksteps * 32;

  const int srow = tid >> 3;                            // staged k row 0..31
  const int c0 = (tid & 7) << 5;                        // staged channel base

  v8f acc[16] = {};

  for (int s = 0; s < ksteps; ++s) {
    const long r = rowbase + (long)s * 32 + srow;
    const float4* p = (const float4*)(x + r * C_DIM + c0);
#pragma unroll
    for (int u = 0; u < 8; ++u) {
      const float4 v = p[u];
      const int c = c0 + (u << 2);
      Cs[c + 0][srow] = (__bf16)v.x;
      Cs[c + 1][srow] = (__bf16)v.y;
      Cs[c + 2][srow] = (__bf16)v.z;
      Cs[c + 3][srow] = (__bf16)v.w;
    }
    if (s + 1 < ksteps)                                  // global_prefetch_b8
      __builtin_prefetch(x + (r + 32) * C_DIM + c0, 0, 1);
    __syncthreads();

    // A 16x32: lanes 0-15 row m K{0..7,16..23}; lanes 16-31 K{8..15,24..31}
    const v8bf a_lo = *(const v8bf*)&Cs[ci + m][half * 8];
    const v8bf a_hi = *(const v8bf*)&Cs[ci + m][half * 8 + 16];
    const v16bf a = cat16(a_lo, a_hi);
#pragma unroll
    for (int g = 0; g < 16; ++g) {
      // B 32x16: lanes 0-15 N=m K{0..7,16..23}; lanes 16-31 K{8..15,24..31}
      const v8bf b_lo = *(const v8bf*)&Cs[(g << 4) + m][half * 8];
      const v8bf b_hi = *(const v8bf*)&Cs[(g << 4) + m][half * 8 + 16];
      acc[g] = wmma_bf16(a, cat16(b_lo, b_hi), acc[g]);
    }
    __syncthreads();
  }

  // C/D layout: VGPR rr -> (M = rr + half*8, N = m)
#pragma unroll
  for (int g = 0; g < 16; ++g)
#pragma unroll
    for (int rr = 0; rr < 8; ++rr)
      atomicAdd(&cov[(ci + half * 8 + rr) * C_DIM + (g << 4) + m], acc[g][rr]);
}

// cov = (cov_raw - N*m_i*m_j) * (1-eps)/(N-1) + eps*I
__global__ __launch_bounds__(256) void fixup_cov_kernel(
    float* __restrict__ cov, const float* __restrict__ mean,
    float scale, float nf) {
  const int i = blockIdx.x, j = threadIdx.x;
  float v = (cov[i * C_DIM + j] - nf * mean[i] * mean[j]) * scale;
  if (i == j) v += EPS_F;
  cov[i * C_DIM + j] = v;
}

// ---------------------------------------------------------------------------
// Pass 3: Cholesky (in place -> L) + per-column forward substitution for
// L^{-1}; emit bf16 wn[i][j] = gamma[i]*Winv[i][j] (row-major = the B operand
// layout the whitening pass wants) and bias[i] = beta[i] - gamma[i]*(Winv m)_i.
// ---------------------------------------------------------------------------
__global__ __launch_bounds__(256) void cholesky_invert_kernel(
    float* __restrict__ A, float* __restrict__ Wi, __bf16* __restrict__ wn,
    const float* __restrict__ gamma, const float* __restrict__ beta,
    const float* __restrict__ mean, float* __restrict__ bias) {
  const int t = threadIdx.x;
  __shared__ float sdiag;

  for (int k = 0; k < C_DIM; ++k) {
    if (t == 0) {
      const float d = sqrtf(A[k * C_DIM + k]);
      A[k * C_DIM + k] = d;
      sdiag = d;
    }
    __syncthreads();
    const float invd = 1.0f / sdiag;
    if (t > k) A[t * C_DIM + k] *= invd;
    __threadfence();
    __syncthreads();
    if (t > k) {
      const float lik = A[t * C_DIM + k];
      for (int j = k + 1; j <= t; ++j)
        A[t * C_DIM + j] -= lik * A[j * C_DIM + k];
    }
    __threadfence();
    __syncthreads();
  }

  // thread t solves L y = e_t (column t of L^{-1}); zero the upper triangle
  {
    const int c = t;
    for (int i = 0; i < c; ++i) Wi[i * C_DIM + c] = 0.0f;
    Wi[c * C_DIM + c] = 1.0f / A[c * C_DIM + c];
    for (int i = c + 1; i < C_DIM; ++i) {
      float s = 0.f;
      for (int j = c; j < i; ++j) s += A[i * C_DIM + j] * Wi[j * C_DIM + c];
      Wi[i * C_DIM + c] = -s / A[i * C_DIM + i];
    }
  }
  __threadfence();
  __syncthreads();

  for (int e = t; e < C_DIM * C_DIM; e += 256) {
    const int i = e >> 8;
    wn[e] = (__bf16)(gamma[i] * Wi[e]);
  }
  float s = 0.f;
  for (int j = 0; j < C_DIM; ++j) s += Wi[t * C_DIM + j] * mean[j];
  bias[t] = beta[t] - gamma[t] * s;
}

// ---------------------------------------------------------------------------
// Pass 4: whitening. out[k][i] = sum_j wn[i][j]*x[k][j] + bias[i].
// One block per 16-row tile: A panel (16x256) staged to LDS ONCE (x read 1x);
// 8 waves x 2 cj-groups cover all 256 output channels. B operand slices are
// contiguous 16B loads from the L2-resident 128KB wn matrix.
// ---------------------------------------------------------------------------
__global__ __launch_bounds__(256) void whiten_kernel(
    const float* __restrict__ x, const __bf16* __restrict__ wn,
    const float* __restrict__ bias, float* __restrict__ out) {
  __shared__ __bf16 As[16][136];     // [row][k] ; 272B row stride (16B mult)

  const int tid = threadIdx.x;
  const int wave = tid >> 5, lane = tid & 31;
  const int half = lane >> 4, m = lane & 15;
  const long r0 = (long)blockIdx.x << 4;

  {  // stage the 16x256 A panel once
    const int row = tid >> 4, c0 = (tid & 15) << 4;
    const float4* p = (const float4*)(x + (r0 + row) * C_DIM + c0);
#pragma unroll
    for (int u = 0; u < 4; ++u) {
      const float4 v = p[u];
      const int c = c0 + (u << 2);
      As[row][c + 0] = (__bf16)v.x;
      As[row][c + 1] = (__bf16)v.y;
      As[row][c + 2] = (__bf16)v.z;
      As[row][c + 3] = (__bf16)v.w;
    }
  }
  __syncthreads();

  v8f acc[2] = {};
  const int cj0 = wave << 5;         // this wave's 32 output channels
#pragma unroll
  for (int s = 0; s < 8; ++s) {
    const int kc = s << 5;
    const v8bf a_lo = *(const v8bf*)&As[m][kc + half * 8];
    const v8bf a_hi = *(const v8bf*)&As[m][kc + half * 8 + 16];
    const v16bf a = cat16(a_lo, a_hi);
#pragma unroll
    for (int g = 0; g < 2; ++g) {
      const __bf16* wrow = wn + (cj0 + (g << 4) + m) * C_DIM + kc + half * 8;
      const v8bf b_lo = *(const v8bf*)wrow;
      const v8bf b_hi = *(const v8bf*)(wrow + 16);
      acc[g] = wmma_bf16(a, cat16(b_lo, b_hi), acc[g]);
    }
  }

#pragma unroll
  for (int g = 0; g < 2; ++g) {
    const int cj = cj0 + (g << 4);
    const float bt = bias[cj + m];
#pragma unroll
    for (int rr = 0; rr < 8; ++rr)
      out[(r0 + half * 8 + rr) * C_DIM + cj + m] = acc[g][rr] + bt;
  }
}

// ---------------------------------------------------------------------------
extern "C" void kernel_launch(void* const* d_in, const int* in_sizes, int n_in,
                              void* d_out, int out_size, void* d_ws, size_t ws_size,
                              hipStream_t stream) {
  const float* x     = (const float*)d_in[0];
  const float* gamma = (const float*)d_in[1];
  const float* beta  = (const float*)d_in[2];
  float* out = (float*)d_out;

  const long N = 64L * 56 * 56;  // 200704 spatial samples

  // Workspace layout (16B aligned):
  char* ws = (char*)d_ws;
  float*  mean = (float*)ws;                            //   1 KB
  float*  cov  = (float*)(ws + 1024);                   // 256 KB
  float*  winv = (float*)(ws + 1024 + 262144);          // 256 KB
  __bf16* wn   = (__bf16*)(ws + 1024 + 2 * 262144);     // 128 KB
  float*  bias = (float*)(ws + 1024 + 2 * 262144 + 131072);  // 1 KB

  // zero mean accumulator + covariance accumulator
  hipMemsetAsync(ws, 0, 1024 + 262144, stream);

  // 1) per-channel mean
  channel_sum_kernel<<<784, 256, 0, stream>>>(x, mean);
  finalize_mean_kernel<<<1, 256, 0, stream>>>(mean, 1.0f / (float)N);

  // 2) raw second moment: grid = (2 ci-halves, 98 K-chunks), 64 ksteps each
  //    (98 * 64 * 32 = 200704 rows); 128 WMMAs per staged 32x256 tile.
  cov_kernel<<<dim3(2, 98), 256, 0, stream>>>(x, cov, 64);

  // 3) mean removal + shrinkage toward identity
  fixup_cov_kernel<<<256, 256, 0, stream>>>(
      cov, mean, (1.0f - EPS_F) / (float)(N - 1), (float)N);

  // 4) Cholesky + triangular inverse + bf16 repack (gamma folded) + bias
  cholesky_invert_kernel<<<1, 256, 0, stream>>>(cov, winv, wn, gamma, beta,
                                                mean, bias);

  // 5) whitening GEMM + bias, NHWC out: 12544 row tiles, x read exactly once
  whiten_kernel<<<12544, 256, 0, stream>>>(x, wn, bias, out);
}
